// VisionAttention_22960895164435
// MI455X (gfx1250) — matrix operand order repaired and checked by reading, not compiled, not run on hardware
//
#include <hip/hip_runtime.h>
#include <cstddef>
#include <cstdint>

typedef __bf16 bf16;
typedef bf16  v8bf  __attribute__((ext_vector_type(8)));
typedef bf16  v16bf __attribute__((ext_vector_type(16)));
typedef float v8f   __attribute__((ext_vector_type(8)));
typedef unsigned int u32x4 __attribute__((ext_vector_type(4)));
typedef int   i32x4 __attribute__((ext_vector_type(4)));
typedef int   i32x8 __attribute__((ext_vector_type(8)));

// Tensor Data Mover availability (device pass only; arity differs by toolchain)
#if defined(__AMDGCN__) && __has_builtin(__builtin_amdgcn_tensor_load_to_lds) && \
    __has_builtin(__builtin_amdgcn_s_wait_tensorcnt)
#define HAVE_TDM 1
#endif

// ---------------------------------------------------------------------------
// helpers
// ---------------------------------------------------------------------------
static __device__ inline bf16 f2bf(float f) {
  union { float f; uint32_t u; } in; in.f = f;
  uint32_t r = (in.u + 0x7FFFu + ((in.u >> 16) & 1u)) >> 16;   // RNE
  union { uint16_t s; bf16 b; } out; out.s = (uint16_t)r;
  return out.b;
}

static __device__ inline v16bf cat8(v8bf lo, v8bf hi) {
  return __builtin_shufflevector(lo, hi, 0,1,2,3,4,5,6,7,8,9,10,11,12,13,14,15);
}

// D = A(16x32) * B(32x16) + C, bf16 in / f32 accum
static __device__ inline v8f wmma_bf16(v16bf a, v16bf b, v8f c) {
  return __builtin_amdgcn_wmma_f32_16x16x32_bf16(false, a, false, b, (short)0, c,
                                                 false, false);
}

// A fragment, 16x32 bf16 tile at `base` (row-major, row stride `ld` elems).
// Layout used here (same K order as B): lane m(0..15) = row m; lanes<16 hold K 0-15, lanes>=16 hold K 16-31.
static __device__ inline v16bf load_a_frag(const bf16* base, int ld, int lane) {
  int row  = lane & 15;
  int koff = (lane >> 4) << 4;                  // 0 or 16
  const bf16* p = base + row * ld + koff;
  return cat8(*(const v8bf*)(p), *(const v8bf*)(p + 8));
}

// B fragment, 32x16: column n = source row n (row-major, K contiguous).
// lanes<16 hold K 0-15, lanes>=16 hold K 16-31. 16B-aligned version (2x b128).
static __device__ inline v16bf load_b_frag16(const bf16* base, int ld, int lane) {
  int col   = lane & 15;
  int khalf = (lane >> 4) << 4;                 // 0 or 16
  const bf16* p = base + col * ld + khalf;
  return cat8(*(const v8bf*)(p), *(const v8bf*)(p + 8));
}

// 32B-aligned version
static __device__ inline v16bf load_b_frag32(const bf16* base, int ld, int lane) {
  int col   = lane & 15;
  int khalf = (lane >> 4) << 4;
  return *(const v16bf*)(base + col * ld + khalf);
}

static __device__ inline bf16 ld_as_bf16(const float* p) { return f2bf(*p); }
static __device__ inline bf16 ld_as_bf16(const bf16*  p) { return *p; }

#if defined(HAVE_TDM)
// Issue a TDM load: 2D tile (tile_d0 x tile_d1 elements, element size per
// g1w0.data_size) from global `gptr` (tensor row stride `stride0` elems) into
// LDS at byte address `lds_addr`. D# packed per CDNA5 ISA ch.8.
static __device__ __attribute__((always_inline)) void
tdm_load(uint32_t lds_addr, const void* gptr, uint32_t g1w0,
         uint32_t tensor_d0, uint32_t tensor_d1,
         uint32_t tile_d0, uint32_t tile_d1, uint64_t stride0) {
  uint64_t ga = (uint64_t)(uintptr_t)gptr;
  u32x4 g0 = { 1u,                                  // count=1, user mode
               lds_addr,                            // lds_addr[31:0]
               (uint32_t)ga,                        // global_addr[31:0]
               (uint32_t)(ga >> 32) | (2u << 30) }; // global_addr[56:32], type=2
  i32x8 g1 = { (int)g1w0,                                        // ds/pad flags
               (int)((tensor_d0 & 0xFFFFu) << 16),               // dim0[15:0]
               (int)((tensor_d0 >> 16) | ((tensor_d1 & 0xFFFFu) << 16)),
               (int)((tensor_d1 >> 16) | (tile_d0 << 16)),
               (int)(tile_d1 & 0xFFFFu),                         // tile_dim1
               (int)(uint32_t)stride0,
               (int)(uint32_t)((stride0 >> 32) & 0xFFFFu),       // stride0 hi
               0 };
  i32x4 z4 = {0, 0, 0, 0};
#if __clang_major__ >= 23
  i32x8 z8 = {0, 0, 0, 0, 0, 0, 0, 0};
  __builtin_amdgcn_tensor_load_to_lds(g0, g1, z4, z4, z8, 0);
#else
  __builtin_amdgcn_tensor_load_to_lds(g0, g1, z4, z4, 0);
#endif
}
#endif

// ---------------------------------------------------------------------------
// Kernel 1/4: C(M x N) = A(M x K) * W(N x K)^T + bias, fp32 out.
// Block = 256 threads (8 wave32), 128x128 output tile, K step 32.
// wave grid 4x2 -> each wave 32x64 via 2x4 WMMA accumulators.
// A-tile staging uses the TDM (with LDS row padding via D#.pad_*) when A is
// already bf16; f32 sources convert on the fly through VALU.
// ---------------------------------------------------------------------------
template <typename AT>
__global__ __launch_bounds__(256)
void gemm_bias_kernel(const AT* __restrict__ A, const float* __restrict__ W,
                      const float* __restrict__ bias, float* __restrict__ C,
                      int N, int K, int NT) {
  __shared__ bf16 As[128 * 40];   // padded stride 40 (16B-aligned rows, bank-skewed)
  __shared__ bf16 Ws[128 * 40];

  const int bN = blockIdx.x % NT;
  const int bM = blockIdx.x / NT;
  const int tid  = threadIdx.x;
  const int lane = tid & 31;
  const int w    = tid >> 5;
  const int wm   = w >> 1;        // 0..3
  const int wn   = w & 1;         // 0..1
  const int lh   = lane >> 4, l15 = lane & 15;

  v8f acc[2][4] = {};

  const int ar = tid >> 1;                 // staging row 0..127
  const int ac = (tid & 1) * 16;           // staging col half
  const AT*    Aptr = A + (size_t)(bM * 128 + ar) * K + ac;
  const float* Wptr = W + (size_t)(bN * 128 + ar) * K + ac;

  for (int k0 = 0; k0 < K; k0 += 32) {
    __syncthreads();

    auto stage_a = [&]() {
#pragma unroll
      for (int j = 0; j < 16; ++j) As[ar * 40 + ac + j] = ld_as_bf16(Aptr + k0 + j);
    };

#if defined(HAVE_TDM)
    if constexpr (sizeof(AT) == 2) {
      // TDM: 32x128 bf16 tile; LDS rows padded 64B data + 16B pad -> stride 40.
      // g1w0: data_size=2B (1<<16), pad_enable (1<<20),
      //       pad_interval=16 DW (3<<22), pad_amount=4 DW (3<<25)
      if (w == 0)
        tdm_load((uint32_t)(uintptr_t)As,
                 A + (size_t)(bM * 128) * K + k0,
                 (1u << 16) | (1u << 20) | (3u << 22) | (3u << 25),
                 (uint32_t)K, 16384u, 32u, 128u, (uint64_t)(uint32_t)K);
    } else {
      stage_a();
    }
#else
    stage_a();
#endif
#pragma unroll
    for (int j = 0; j < 16; ++j) Ws[ar * 40 + ac + j] = f2bf(Wptr[k0 + j]);
    if (k0 + 32 < K) {
      __builtin_prefetch(Aptr + k0 + 32, 0, 1);   // global_prefetch_b8
      __builtin_prefetch(Wptr + k0 + 32, 0, 1);
    }
#if defined(HAVE_TDM)
    if constexpr (sizeof(AT) == 2) {
      if (w == 0) __builtin_amdgcn_s_wait_tensorcnt((short)0);
    }
#endif
    __syncthreads();

    v16bf bfrg[4];
#pragma unroll
    for (int nt = 0; nt < 4; ++nt)
      bfrg[nt] = load_b_frag16(Ws + (wn * 64 + nt * 16) * 40, 40, lane);
#pragma unroll
    for (int mt = 0; mt < 2; ++mt) {
      v16bf afrg = load_a_frag(As + (wm * 32 + mt * 16) * 40, 40, lane);
#pragma unroll
      for (int nt = 0; nt < 4; ++nt)
        acc[mt][nt] = wmma_bf16(afrg, bfrg[nt], acc[mt][nt]);
    }
  }

  // epilogue: C-layout = lane l holds col (l&15), rows i + 8*(l>>4)
#pragma unroll
  for (int mt = 0; mt < 2; ++mt)
#pragma unroll
    for (int nt = 0; nt < 4; ++nt) {
      int col = bN * 128 + wn * 64 + nt * 16 + l15;
      float bv = bias[col];
#pragma unroll
      for (int i = 0; i < 8; ++i) {
        int row = bM * 128 + wm * 32 + mt * 16 + i + 8 * lh;
        C[(size_t)row * N + col] = acc[mt][nt][i] + bv;
      }
    }
}

// ---------------------------------------------------------------------------
// Kernel 2/4: rotary(q,k) + fold 1/sqrt(80) into q + layout:
//   Qp,Kp: (n,h,576,96) bf16  (D padded 80->96 with zeros)
//   Vt:    (n,h,96,576) bf16  (transposed, padded rows zero)
// ---------------------------------------------------------------------------
__global__ __launch_bounds__(256)
void rotary_kernel(const float* __restrict__ qkv, const float* __restrict__ freqs,
                   bf16* __restrict__ Qp, bf16* __restrict__ Kp,
                   bf16* __restrict__ Vt) {
  size_t gid = (size_t)blockIdx.x * 256 + threadIdx.x;   // exactly 9216*16*96
  int d   = (int)(gid % 96);
  int h   = (int)((gid / 96) % 16);
  int tok = (int)(gid / (96 * 16));
  int n = tok / 576, l = tok % 576;

  float qv = 0.f, kv = 0.f, vv = 0.f;
  size_t base = (size_t)tok * 3840 + h * 80;
  if (d < 80) {
    float q = qkv[base + d];
    float k = qkv[base + 1280 + d];
    vv      = qkv[base + 2560 + d];
    int dr  = (d < 40) ? d : d - 40;
    int dp  = (d < 40) ? d + 40 : d - 40;
    float fr = freqs[(size_t)tok * 40 + dr];
    float c = __cosf(fr), s = __sinf(fr);
    float qo = qkv[base + dp];
    float ko = qkv[base + 1280 + dp];
    qv = (d < 40) ? (q * c - qo * s) : (q * c + qo * s);
    kv = (d < 40) ? (k * c - ko * s) : (k * c + ko * s);
    qv *= 0.11180339887498949f;   // 1/sqrt(80)
  }
  size_t nh = (size_t)(n * 16 + h);
  Qp[nh * 576 * 96 + (size_t)l * 96 + d] = f2bf(qv);
  Kp[nh * 576 * 96 + (size_t)l * 96 + d] = f2bf(kv);
  Vt[nh * 96 * 576 + (size_t)d * 576 + l] = f2bf(vv);
}

// ---------------------------------------------------------------------------
// Kernel 3/4: block-diagonal attention, one WG per (n,h,qtile of 64 queries).
// Scores 64x576 f32 in LDS -> softmax -> bf16 P in LDS -> P * V^T via WMMA.
// Uses ~240KB of the 320KB WGP LDS (dynamic shared). Q tile comes in via TDM.
// ---------------------------------------------------------------------------
#define SROW 592   // 576 + 16 pad (bank skew), multiple of 8

__global__ __launch_bounds__(256)
void attn_kernel(const bf16* __restrict__ Qp, const bf16* __restrict__ Kp,
                 const bf16* __restrict__ Vt, bf16* __restrict__ AO) {
  extern __shared__ char dyn[];
  float* Sc = (float*)dyn;                           // 64 x SROW f32 scores
  bf16*  Ps = (bf16*)(dyn + (size_t)64 * SROW * 4);  // 64 x SROW bf16 probs
  __shared__ bf16  Qs[64 * 96];
  __shared__ float rowm[64], rowinv[64], red[256];

  const int b  = blockIdx.x;        // (n*16+h)*9 + qt
  const int qt = b % 9;
  const int nh = b / 9;
  const int q0 = qt * 64;
  const bf16* Qph = Qp + (size_t)nh * 576 * 96;
  const bf16* Kph = Kp + (size_t)nh * 576 * 96;
  const bf16* Vth = Vt + (size_t)nh * 96 * 576;

  const int tid = threadIdx.x, lane = tid & 31, w = tid >> 5;
  const int lh = lane >> 4, l15 = lane & 15;

  // load Q tile 64x96 bf16 (contiguous 12KB) into LDS
#if defined(HAVE_TDM)
  if (w == 0) {
    // 1-D copy: 3072 DWORDs (data_size=4B -> 2<<16)
    tdm_load((uint32_t)(uintptr_t)Qs, Qph + (size_t)q0 * 96,
             (2u << 16), 3072u, 1u, 3072u, 1u, 3072ull);
    __builtin_amdgcn_s_wait_tensorcnt((short)0);
  }
#else
  for (int ch = tid; ch < 768; ch += 256) {
    int r = ch / 12, c = (ch % 12) * 8;
    *(v8bf*)(Qs + r * 96 + c) = *(const v8bf*)(Qph + (size_t)(q0 + r) * 96 + c);
  }
#endif
  __syncthreads();

  // ---- scores: S = Q * K^T (K-dim = 96 = 3 x 32), 36 N-tiles over 8 waves
  for (int nb = w; nb < 36; nb += 8) {
    v8f acc[4] = {};
#pragma unroll
    for (int c = 0; c < 3; ++c) {
      v16bf bfrg = load_b_frag32(Kph + (size_t)(nb * 16) * 96 + c * 32, 96, lane);
#pragma unroll
      for (int mt = 0; mt < 4; ++mt) {
        v16bf afrg = load_a_frag(Qs + mt * 16 * 96 + c * 32, 96, lane);
        acc[mt] = wmma_bf16(afrg, bfrg, acc[mt]);
      }
    }
#pragma unroll
    for (int mt = 0; mt < 4; ++mt)
#pragma unroll
      for (int i = 0; i < 8; ++i)
        Sc[(mt * 16 + i + 8 * lh) * SROW + nb * 16 + l15] = acc[mt][i];
  }
  __syncthreads();

  // ---- softmax over 576 cols: 4 threads per row, 144 cols each
  {
    const int row = tid >> 2, part = tid & 3;
    const float* sr = Sc + row * SROW + part * 144;
    float m = -1e30f;
    for (int j = 0; j < 144; ++j) m = fmaxf(m, sr[j]);
    red[tid] = m;
    __syncthreads();
    if (part == 0)
      rowm[row] = fmaxf(fmaxf(red[tid], red[tid + 1]),
                        fmaxf(red[tid + 2], red[tid + 3]));
    __syncthreads();
    float mm = rowm[row];
    bf16* pr = Ps + row * SROW + part * 144;
    float s = 0.f;
    for (int j = 0; j < 144; ++j) {
      float p = __expf(sr[j] - mm);
      s += p;
      pr[j] = f2bf(p);
    }
    red[tid] = s;
    __syncthreads();
    if (part == 0)
      rowinv[row] = 1.f / (red[tid] + red[tid + 1] + red[tid + 2] + red[tid + 3]);
    __syncthreads();
  }

  // ---- O = P * V^T : 24 (16x16) tiles (4 M x 6 N) -> 3 per wave
  const int mt  = w >> 1;            // 0..3
  const int ntb = (w & 1) * 3;       // 0 or 3
  v8f acc[3] = {};
  for (int kb = 0; kb < 576; kb += 32) {
    v16bf afrg = load_a_frag(Ps + (size_t)(mt * 16) * SROW + kb, SROW, lane);
#pragma unroll
    for (int j = 0; j < 3; ++j) {
      v16bf bfrg =
          load_b_frag32(Vth + (size_t)((ntb + j) * 16) * 576 + kb, 576, lane);
      acc[j] = wmma_bf16(afrg, bfrg, acc[j]);
    }
  }
  const int n = nh >> 4, h = nh & 15;
#pragma unroll
  for (int j = 0; j < 3; ++j) {
    int nt = ntb + j;
    int d  = nt * 16 + l15;
    if (d < 80) {                          // drop the zero-padded d=80..95 tile
#pragma unroll
      for (int i = 0; i < 8; ++i) {
        int r = mt * 16 + i + 8 * lh;
        float v = acc[j][i] * rowinv[r];
        size_t token = (size_t)n * 576 + q0 + r;
        AO[token * 1280 + h * 80 + d] = f2bf(v);
      }
    }
  }
}

// ---------------------------------------------------------------------------
// launcher
// ---------------------------------------------------------------------------
extern "C" void kernel_launch(void* const* d_in, const int* in_sizes, int n_in,
                              void* d_out, int out_size, void* d_ws, size_t ws_size,
                              hipStream_t stream) {
  (void)in_sizes; (void)n_in; (void)out_size; (void)ws_size;
  const float* x     = (const float*)d_in[0];
  const float* rot   = (const float*)d_in[1];
  const float* w_qkv = (const float*)d_in[2];
  const float* b_qkv = (const float*)d_in[3];
  const float* w_out = (const float*)d_in[4];
  const float* b_out = (const float*)d_in[5];
  // d_in[6] cu_seqlens: static equal-length blocks (16 x 576), honored by layout.

  char* ws = (char*)d_ws;
  size_t off = 0;
  float* qkvws = (float*)(ws + off); off += (size_t)9216 * 3840 * 4;
  bf16*  Qp    = (bf16*) (ws + off); off += (size_t)16 * 16 * 576 * 96 * 2;
  bf16*  Kp    = (bf16*) (ws + off); off += (size_t)16 * 16 * 576 * 96 * 2;
  bf16*  Vt    = (bf16*) (ws + off); off += (size_t)16 * 16 * 96 * 576 * 2;
  bf16*  AO    = (bf16*) (ws + off);

  // 1) QKV projection: (9216x1280) x (3840x1280)^T + b_qkv -> fp32
  gemm_bias_kernel<float><<<72 * 30, 256, 0, stream>>>(x, w_qkv, b_qkv, qkvws,
                                                       3840, 1280, 30);
  // 2) rotary + attention layouts
  rotary_kernel<<<55296, 256, 0, stream>>>(qkvws, rot, Qp, Kp, Vt);

  // 3) attention (needs >64KB dynamic LDS; WGP has 320KB)
  size_t smem = (size_t)64 * SROW * 4 + (size_t)64 * SROW * 2;
  hipFuncSetAttribute((const void*)attn_kernel,
                      hipFuncAttributeMaxDynamicSharedMemorySize, (int)smem);
  attn_kernel<<<16 * 16 * 9, 256, smem, stream>>>(Qp, Kp, Vt, AO);

  // 4) output projection: (9216x1280 bf16) x (1280x1280)^T + b_out -> d_out
  gemm_bias_kernel<bf16><<<72 * 10, 256, 0, stream>>>(AO, w_out, b_out,
                                                      (float*)d_out, 1280, 1280, 10);
}
